// NNconv_5738076308179
// MI455X (gfx1250) — matrix operand order repaired and checked by reading, not compiled, not run on hardware
//
#include <hip/hip_runtime.h>

// ---------------------------------------------------------------------------
// NNConv (edge-conditioned conv) fused for MI455X / gfx1250, wave32 + WMMA.
//
// msg = G @ W2v  where G[e, k*128+i] = h[e,k]*x[src[e],i]  (outer product,
// built in registers), W2v = w2 reinterpreted [16384,128], plus 128 virtual
// K-rows (h==1) carrying b2. B double-buffered in registers with
// sched_barrier-enforced software pipelining; K split 4-way across blocks.
// ---------------------------------------------------------------------------

#define DD   128
#define EE   16384
#define NNODES 10000
#define KCHUNKS 516                         // (16384 + 128) / 32
#define BP_ELEMS (KCHUNKS * 8 * 32 * 16)    // 2,113,536 bf16 (real data)

typedef __attribute__((ext_vector_type(16))) __bf16 v16bf;
typedef __attribute__((ext_vector_type(8)))  float  v8f;

__device__ __forceinline__ void atomAddF(float* p, float v) {
    __hip_atomic_fetch_add(p, v, __ATOMIC_RELAXED, __HIP_MEMORY_SCOPE_AGENT);
}

// --- k0: pack w2 (+ b2 as rows 16384..16511) into WMMA-B lane layout --------
// Bp[((c*8+t)*32 + L)*16 + j] = bf16( W2v[c*32 + kB(L) + j, t*16 + (L&15)] )
__global__ __launch_bounds__(256) void k0_pack(const float* __restrict__ w2,
                                               const float* __restrict__ b2,
                                               __bf16* __restrict__ Bp) {
    int tid = blockIdx.x * 256 + threadIdx.x;
    if (tid >= BP_ELEMS) return;
    int j  = tid & 15;
    int L  = (tid >> 4) & 31;
    int t  = (tid >> 9) & 7;
    int c  = tid >> 12;
    int kB = L & 16;                 // 0 or 16
    int col = t * 16 + (L & 15);
    int rl  = kB + j;                // 0..31 within chunk
    float v;
    if (c < 512) v = w2[(size_t)(c * 32 + rl) * 128 + col];   // W2v[r,o]=w2_flat[r*128+o]
    else         v = b2[((c - 512) * 32 + rl) * 128 + col];   // b2 as [i,o]
    Bp[tid] = (__bf16)v;
}

// --- k1: h = relu(edge_attr @ w1 + b1), stored K-major h_t[k*EE + e] --------
__global__ __launch_bounds__(256) void k1_h(const float* __restrict__ ea,
                                            const float* __restrict__ w1,
                                            const float* __restrict__ b1,
                                            float* __restrict__ h_t) {
    __shared__ __bf16 eat[128 * 132];            // transposed tile, padded
    const int u = threadIdx.x;
    const int eBase = blockIdx.x * 128;
    for (int idx = u; idx < 128 * 128; idx += 256) {
        int el = idx >> 7, k = idx & 127;
        eat[k * 132 + el] = (__bf16)ea[(size_t)(eBase + el) * 128 + k];
    }
    __syncthreads();
    const int el = u & 127;
    const int oh = u >> 7;                       // wave-uniform -> scalar w1 loads
    for (int oo = 0; oo < 64; ++oo) {
        const int o = oh * 64 + oo;
        float s = 0.f;
        #pragma unroll 8
        for (int k = 0; k < 128; ++k)
            s += (float)eat[k * 132 + el] * w1[k * 128 + o];
        s += b1[o];
        h_t[(size_t)o * EE + eBase + el] = fmaxf(s, 0.f);   // coalesced, transposed
    }
}

// --- k2 helper: one 32-wide K chunk -----------------------------------------
// Loads next chunk's 8 B tiles (immediate offsets off a strided base), then a
// sched_barrier(0) pins them above the compute; WMMAs consume bcur.  Loads of
// chunk c+1 may hoist among chunk c's WMMAs but can never sink to their
// consumers, so bv0/bv1 stay distinct and >=8 WMMAs separate issue from wait.
__device__ __forceinline__ void chunk_step(float hv, const float* xr16,
                                           const char* bk, int onext,
                                           v16bf (&bcur)[8], v16bf (&bnext)[8],
                                           v8f (&acc)[8]) {
    #pragma unroll
    for (int t = 0; t < 8; ++t)
        bnext[t] = *(const v16bf*)(bk + onext + t * 1024);
    __builtin_amdgcn_sched_barrier(0);
    v16bf av;
    #pragma unroll
    for (int j = 0; j < 16; ++j)
        av[j] = (__bf16)(hv * xr16[j]);           // G tile on the fly
    #pragma unroll
    for (int t = 0; t < 8; ++t)
        acc[t] = __builtin_amdgcn_wmma_f32_16x16x32_bf16(
                     false, av, false, bcur[t], (short)0, acc[t], false, false);
}

// --- k2: WMMA GEMM, K split 4 ways; writes partial msg (no atomics) --------
// grid (128, 4): x = edge block (128 edges, 8 waves x 16 edges), y = K split.
__global__ __launch_bounds__(256) void k2_msg(const float* __restrict__ x,
                                              const int*   __restrict__ ei,
                                              const float* __restrict__ h_t,
                                              const __bf16* __restrict__ Bp,
                                              float* __restrict__ msg) {
    const int u    = threadIdx.x;
    const int wave = u >> 5;
    const int lane = u & 31;
    const int s    = blockIdx.y;                  // K split id
    const int kr0  = s * 32;
    const int kr1  = (s == 3) ? 129 : kr0 + 32;   // split 3 owns virtual b2 row
    float* msgp = msg + (size_t)s * EE * DD;

    const int eBase = blockIdx.x * 128 + wave * 16;
    const int m  = lane & 15;                     // A-row M within tile
    const int kA = (lane < 16) ? 0 : 16;          // K-half base (contiguous order k = 16h + j)
    const int e  = eBase + m;
    const float* xs = x + (size_t)ei[e] * DD;     // gather source node row

    // Pre-gather exactly the 64 x-values this lane's A slots use.
    float xr[64];
    #pragma unroll
    for (int q = 0; q < 4; ++q) {
        float4 a0 = *(const float4*)(xs + q * 32 + kA);
        float4 a1 = *(const float4*)(xs + q * 32 + kA + 4);
        float4 c0 = *(const float4*)(xs + q * 32 + 8 + kA);
        float4 c1 = *(const float4*)(xs + q * 32 + 12 + kA);
        xr[q*16+ 0]=a0.x; xr[q*16+ 1]=a0.y; xr[q*16+ 2]=a0.z; xr[q*16+ 3]=a0.w;
        xr[q*16+ 4]=a1.x; xr[q*16+ 5]=a1.y; xr[q*16+ 6]=a1.z; xr[q*16+ 7]=a1.w;
        xr[q*16+ 8]=c0.x; xr[q*16+ 9]=c0.y; xr[q*16+10]=c0.z; xr[q*16+11]=c0.w;
        xr[q*16+12]=c1.x; xr[q*16+13]=c1.y; xr[q*16+14]=c1.z; xr[q*16+15]=c1.w;
    }

    v8f acc[8];
    #pragma unroll
    for (int t = 0; t < 8; ++t) acc[t] = (v8f){0.f,0.f,0.f,0.f,0.f,0.f,0.f,0.f};

    // Strided per-lane base: advances 4 chunks (32KB) per k-row, so all inner
    // loads are immediate-offset forms off one pointer (no 64-bit re-derives).
    const char* bk = (const char*)(Bp + lane * 16) + (size_t)kr0 * 4 * 8192;
    const float* hp = h_t + kr0 * EE + e;

    // Prologue: load this split's first chunk into buffer 0.
    v16bf bv0[8], bv1[8];
    #pragma unroll
    for (int t = 0; t < 8; ++t)
        bv0[t] = *(const v16bf*)(bk + t * 1024);

    for (int kr = kr0; kr < kr1; ++kr) {
        const float hv = (kr < 128) ? *hp : 1.0f;
        hp += EE;
        __builtin_prefetch(bk + 8 * 8192, 0, 3);  // warm ~64KB ahead
        chunk_step(hv, &xr[ 0], bk, 1 * 8192, bv0, bv1, acc);
        chunk_step(hv, &xr[16], bk, 2 * 8192, bv1, bv0, acc);
        chunk_step(hv, &xr[32], bk, 3 * 8192, bv0, bv1, acc);
        chunk_step(hv, &xr[48], bk, 4 * 8192, bv1, bv0, acc);
        bk += 4 * 8192;
    }

    // C layout: element v -> M = v + 8*(lane>>4), N = t*16 + (lane&15).
    const int ehi = 8 * (lane >> 4);
    #pragma unroll
    for (int t = 0; t < 8; ++t) {
        const int col = t * 16 + m;
        #pragma unroll
        for (int v = 0; v < 8; ++v)
            msgp[(size_t)(eBase + v + ehi) * DD + col] = acc[t][v];
    }
}

// --- k2r: reduce 4 K-split partials, scatter-mean into seg/cnt -------------
__global__ __launch_bounds__(256) void k2r(const float* __restrict__ msg,
                                           const int* __restrict__ ei,
                                           float* __restrict__ seg,
                                           unsigned int* __restrict__ cnt) {
    const size_t tid = (size_t)blockIdx.x * 256 + threadIdx.x;  // 0 .. EE*DD-1
    const int e = (int)(tid >> 7);
    const int o = (int)(tid & 127);
    const size_t S = (size_t)EE * DD;
    float v = msg[tid] + msg[tid + S] + msg[tid + 2 * S] + msg[tid + 3 * S];
    const int d = ei[EE + e];
    if (o == 0) atomicAdd(cnt + d, 1u);
    atomAddF(seg + (size_t)d * DD + o, v);
}

// --- k3: out = x + gelu(seg/max(cnt,1) + x@root + bias) --------------------
__global__ __launch_bounds__(256) void k3_out(const float* __restrict__ x,
                                              const float* __restrict__ root,
                                              const float* __restrict__ bias,
                                              const float* __restrict__ seg,
                                              const unsigned int* __restrict__ cnt,
                                              float* __restrict__ out) {
    __shared__ __bf16 xt[128 * 132];
    const int u = threadIdx.x;
    const int nBase = blockIdx.x * 128;
    for (int idx = u; idx < 128 * 128; idx += 256) {
        int nl = idx >> 7, k = idx & 127;
        int n = nBase + nl;
        xt[k * 132 + nl] = (__bf16)((n < NNODES) ? x[(size_t)n * 128 + k] : 0.f);
    }
    __syncthreads();
    const int nl = u & 127;
    const int n  = nBase + nl;
    if (n >= NNODES) return;
    const int oh = u >> 7;
    const float cf = fmaxf((float)cnt[n], 1.0f);
    for (int oo = 0; oo < 64; ++oo) {
        const int o = oh * 64 + oo;
        float s = 0.f;
        #pragma unroll 8
        for (int k = 0; k < 128; ++k)
            s += (float)xt[k * 132 + nl] * root[k * 128 + o];
        float v = seg[(size_t)n * 128 + o] / cf + s + bias[o];
        float g = 0.5f * v * (1.0f + erff(v * 0.70710678118654752440f));
        out[(size_t)n * 128 + o] = x[(size_t)n * 128 + o] + g;   // exact residual
    }
}

// --- launch -----------------------------------------------------------------
extern "C" void kernel_launch(void* const* d_in, const int* in_sizes, int n_in,
                              void* d_out, int out_size, void* d_ws, size_t ws_size,
                              hipStream_t stream) {
    const float* x    = (const float*)d_in[0];
    const int*   ei   = (const int*)d_in[1];     // edge_index [2,E]
    const float* ea   = (const float*)d_in[2];
    const float* w1   = (const float*)d_in[3];
    const float* b1   = (const float*)d_in[4];
    const float* w2   = (const float*)d_in[5];
    const float* b2   = (const float*)d_in[6];
    const float* root = (const float*)d_in[7];
    const float* bias = (const float*)d_in[8];
    float* out = (float*)d_out;

    char* ws = (char*)d_ws;
    float*        h_t = (float*)ws;                       //  8,388,608 B
    __bf16*       Bp  = (__bf16*)(ws + 8388608);          //  4,268,032 B (padded)
    float*        msg = (float*)(ws + 12656640);          // 33,554,432 B (4 partials)
    float*        seg = (float*)(ws + 46211072);          //  5,120,000 B
    unsigned int* cnt = (unsigned int*)(ws + 51331072);   //     40,000 B

    hipMemsetAsync(ws + 46211072, 0, 5160000, stream);    // seg + cnt

    k0_pack<<<(BP_ELEMS + 255) / 256, 256, 0, stream>>>(w2, b2, Bp);
    k1_h  <<<EE / 128, 256, 0, stream>>>(ea, w1, b1, h_t);
    k2_msg<<<dim3(EE / 128, 4), 256, 0, stream>>>(x, ei, h_t, Bp, msg);
    k2r   <<<(EE * DD) / 256, 256, 0, stream>>>(msg, ei, seg, cnt);
    k3_out<<<(NNODES + 127) / 128, 256, 0, stream>>>(x, root, bias, seg, cnt, out);
}
